// ModulatedConv2d_76338748719240
// MI455X (gfx1250) — matrix operand order repaired and checked by reading, not compiled, not run on hardware
//
#include <hip/hip_runtime.h>

typedef __bf16 bf16_t;
typedef bf16_t v16bf __attribute__((ext_vector_type(16)));
typedef float  v8f   __attribute__((ext_vector_type(8)));
typedef int    v4i_t __attribute__((vector_size(16)));
typedef unsigned int u32x4 __attribute__((vector_size(16)));
typedef int    i32x8 __attribute__((vector_size(32)));
typedef int    i32x4 __attribute__((vector_size(16)));

#define IN_CH   512
#define OUT_CH  512
#define HW      4096   // 64*64
#define KC      32
#define ATAP    (256 * KC)     // one tap's A tile: 256 rows x 32 k = 8192 elems (16 KB)
#define NSTAGES 48             // (512/32) * 3 tap-rows

#if __has_builtin(__builtin_amdgcn_tensor_load_to_lds) && \
    __has_builtin(__builtin_amdgcn_s_wait_tensorcnt)
#define USE_TDM 1
#else
#define USE_TDM 0
#endif
#if __has_builtin(__builtin_amdgcn_global_load_async_to_lds_b128) && \
    __has_builtin(__builtin_amdgcn_s_wait_asynccnt)
#define USE_ASYNC_LDS 1
#else
#define USE_ASYNC_LDS 0
#endif

#if USE_TDM
#define WAIT_A_PIPE() __builtin_amdgcn_s_wait_tensorcnt(3)   // 3 TDM ops per stage
#define WAIT_A_ALL()  __builtin_amdgcn_s_wait_tensorcnt(0)
#elif USE_ASYNC_LDS
#define WAIT_A_PIPE() __builtin_amdgcn_s_wait_asynccnt(12)   // 12 async loads/thread/stage
#define WAIT_A_ALL()  __builtin_amdgcn_s_wait_asynccnt(0)
#else
#define WAIT_A_PIPE()
#define WAIT_A_ALL()
#endif

// -------------------- kernel 1: s = style @ mod_w^T  [8,512] --------------------
__global__ __launch_bounds__(256) void compute_s_kernel(
    const float* __restrict__ style, const float* __restrict__ mod_w,
    float* __restrict__ s) {
  int idx = blockIdx.x * 256 + threadIdx.x;    // 4096 = 8*512
  int b = idx >> 9, i = idx & 511;
  const float4* st = (const float4*)(style + (size_t)b * 512);
  const float4* mw = (const float4*)(mod_w + (size_t)i * 512);
  float acc = 0.f;
  #pragma unroll 4
  for (int d = 0; d < 128; ++d) {
    float4 a = st[d], w = mw[d];
    acc += a.x * w.x + a.y * w.y + a.z * w.z + a.w * w.w;
  }
  s[idx] = acc;
}

// ---------- kernel 2: oscale[b,cout] = scale * rsqrt(scale^2 * sum + 1e-8) ----------
__global__ __launch_bounds__(256) void compute_oscale_kernel(
    const float* __restrict__ weight, const float* __restrict__ s,
    float* __restrict__ oscale) {
  int idx = blockIdx.x * 256 + threadIdx.x;    // 4096 = 8*512
  int b = idx >> 9, co = idx & 511;
  const float* sb = s + (size_t)b * 512;
  float acc = 0.f;
  for (int ci = 0; ci < 512; ++ci) {
    const float* wp = weight + ((size_t)co * 512 + ci) * 9;
    float wsq = 0.f;
    #pragma unroll
    for (int t = 0; t < 9; ++t) { float w = wp[t]; wsq += w * w; }
    float sv = sb[ci];
    acc += wsq * sv * sv;
  }
  const float kScale = 0.014731391274719739f;  // 1/sqrt(512*9)
  oscale[idx] = kScale * rsqrtf(kScale * kScale * acc + 1e-8f);
}

// ---- kernel 3: wbf[kk][cout][cin'] = bf16(weight[cout][cin][kk]), cin' applies the
//      A-fragment 8-element K-group swap so the LDS copy can be a linear DMA ----
__global__ __launch_bounds__(256) void transpose_w_kernel(
    const float* __restrict__ weight, bf16_t* __restrict__ wbf) {
  int idx = blockIdx.x * 256 + threadIdx.x;    // 9*512*512
  int cin = idx & 511;
  int t   = idx >> 9;
  int cout = t & 511;
  int kk   = t >> 9;                           // 0..8
  int grp  = (cin >> 3) & 3;                   // K-group within 32-block
  int cind = (grp == 1) ? cin + 8 : (grp == 2) ? cin - 8 : cin;
  wbf[((size_t)t << 9) | cind] =
      (bf16_t)weight[((size_t)cout * 512 + cin) * 9 + kk];
}

#if USE_TDM
// One 2D TDM descriptor: tile 32 (cin, 2B elems) x 256 (cout rows),
// global row stride 512 elems, contiguous into LDS (16 KB).
__device__ __forceinline__ void tdm_issue_A(const bf16_t* gsrc, unsigned lds_off) {
  unsigned long long ga = (unsigned long long)gsrc;
  u32x4 g0;
  g0[0] = 1u;                                                // count=1 (valid), user
  g0[1] = lds_off;                                           // LDS byte address
  g0[2] = (unsigned)ga;                                      // global_addr[31:0]
  g0[3] = (unsigned)((ga >> 32) & 0x01FFFFFFu) | (2u << 30); // addr[56:32] | type=2
  i32x8 g1;
  g1[0] = 1 << 16;             // workgroup_mask=0 | data_size=1 (2 bytes)
  g1[1] = (int)(512u << 16);   // atomic_barrier_addr=0 | tensor_dim0 lo16 = 512
  g1[2] = (int)(512u << 16);   // tensor_dim0 hi16=0   | tensor_dim1 lo16 = 512
  g1[3] = (int)(32u << 16);    // tensor_dim1 hi16=0   | tile_dim0 = 32
  g1[4] = 256;                 // tile_dim1 = 256 | tile_dim2 = 0
  g1[5] = 512;                 // tensor_dim0_stride lo32 = 512 elems
  g1[6] = 0;                   // stride hi16 | tensor_dim1_stride lo16
  g1[7] = 0;
  i32x4 z4 = {0, 0, 0, 0};
  i32x8 z8 = {0, 0, 0, 0, 0, 0, 0, 0};
  __builtin_amdgcn_tensor_load_to_lds(g0, g1, z4, z4, z8, 0);
}
#endif

// -------------------- kernel 4: implicit-GEMM conv with bf16 WMMA --------------------
// Block tile M=256 x N=128; 8 waves as 4(M) x 2(N); wave tile 64x64.
// Pipeline stage = one tap-row (kh fixed, 3 taps, K-depth 96): 3 TDM descriptors
// fill a 48 KB A buffer (double-buffered); 48 WMMAs per wave per barrier pair.
// B halo tile [4 rows][66 cols][32 cin] staged once per cin-chunk.
__global__ __launch_bounds__(256) void conv_wmma_kernel(
    const float* __restrict__ x,        // [8,512,64,64]
    const bf16_t* __restrict__ wbf,     // [9,512,512] (k-group pre-permuted)
    const float* __restrict__ s,        // [8,512]
    const float* __restrict__ oscale,   // [8,512]
    float* __restrict__ out) {          // [8,512,64,64]
  __shared__ __attribute__((aligned(64))) bf16_t Alds[2][3 * ATAP];    // 96 KB
  __shared__ __attribute__((aligned(64))) bf16_t Bhalo[4 * 66 * KC];   // 16.5 KB

  const int tid   = threadIdx.x;
  const int lane  = tid & 31;
  const int l16   = lane & 15;
  const int khalf = lane >> 4;
  const int wv    = tid >> 5;     // 0..7
  const int wm    = wv & 3;       // 4 M-waves
  const int wn    = wv >> 2;      // 2 N-waves

  int bid = blockIdx.x;                  // 512 = 8 * 2 * 32
  const int nt = bid & 31;  bid >>= 5;
  const int mt = bid & 1;   bid >>= 1;
  const int b  = bid;

  const int m0 = mt * 256;
  const int p0 = nt * 128;
  const int h0 = p0 >> 6;

  // per-fragment N coordinates (independent of tap)
  int hl[4], wc[4];
  #pragma unroll
  for (int nf = 0; nf < 4; ++nf) {
    int n = wn * 64 + nf * 16 + l16;
    hl[nf] = n >> 6;
    wc[nf] = n & 63;
  }

  v8f acc[4][4];
  #pragma unroll
  for (int i = 0; i < 4; ++i)
    #pragma unroll
    for (int j = 0; j < 4; ++j)
      acc[i][j] = (v8f){0.f, 0.f, 0.f, 0.f, 0.f, 0.f, 0.f, 0.f};

  const float* xb = x + (size_t)b * IN_CH * HW;
  const float* sb = s + (size_t)b * IN_CH;

  // B staging mapping: cin = tid>>3, image row = (tid&7)>>1, 33-col strip = tid&1
  const int bc    = tid >> 3;
  const int brow  = (tid & 7) >> 1;
  const int bhalf = tid & 1;

  // A stager: stage = (c0/32)*3 + r; fills taps kk = 3r .. 3r+2 for cin-chunk c0
  auto stageA = [&](int stage, int buf) {
    int cc = stage / 3;
    int r  = stage - cc * 3;
    #pragma unroll
    for (int t = 0; t < 3; ++t) {
      int kk = 3 * r + t;
      const bf16_t* src = wbf + ((size_t)kk * OUT_CH + m0) * IN_CH + cc * KC;
      bf16_t* dbase = &Alds[buf][t * ATAP];
#if USE_TDM
      if (wv == 0)
        tdm_issue_A(src, (unsigned)(unsigned long long)(void*)dbase);
#else
      #pragma unroll
      for (int it = 0; it < 4; ++it) {
        int ch  = tid + it * 256;        // 1024 x 16B chunks, linear
        int row = ch >> 2, j = ch & 3;
        const bf16_t* p = src + (size_t)row * IN_CH + j * 8;
        bf16_t*       d = dbase + ch * 8;
  #if USE_ASYNC_LDS
        __builtin_amdgcn_global_load_async_to_lds_b128((v4i_t*)p, (v4i_t*)d, 0, 0);
  #else
        *(uint4*)d = *(const uint4*)p;
  #endif
      }
#endif
    }
  };

  stageA(0, 0);                          // prologue: first tap-row in flight

  int stage = 0;
  for (int c0 = 0; c0 < IN_CH; c0 += KC) {
    // ---- stage B halo tile once per cin-chunk (prev compute ended with barrier) ----
    {
      const float  sv = sb[c0 + bc];
      const float* xc = xb + (size_t)(c0 + bc) * HW;
      const int y = h0 - 1 + brow;
      const bool yok = (unsigned)y < 64u;
      const float* xrow = xc + y * 64;
      for (int i = 0; i < 33; ++i) {
        int col = bhalf * 33 + i;        // 0..65
        int xx  = col - 1;
        float v = 0.f;
        if (yok && (unsigned)xx < 64u) v = xrow[xx] * sv;
        Bhalo[(brow * 66 + col) * KC + (((bc >> 3) & 3) == 1 ? bc + 8 : ((bc >> 3) & 3) == 2 ? bc - 8 : bc)] = (bf16_t)v;
      }
    }

    for (int r = 0; r < 3; ++r, ++stage) {     // kh = r
      // pipeline: issue next tap-row, then wait for the current one (in-order TDM)
      if (stage + 1 < NSTAGES) {
        stageA(stage + 1, (stage + 1) & 1);
        WAIT_A_PIPE();
      } else {
        WAIT_A_ALL();
      }
      __syncthreads();                   // A(stage) + B writes visible to all waves

      const bf16_t* Acur = &Alds[stage & 1][0];

      // ---- 3 taps x 16 = 48 WMMAs per wave per barrier pair ----
      #pragma unroll
      for (int t = 0; t < 3; ++t) {      // kw = t
        const bf16_t* At = Acur + t * ATAP;
        v16bf bfrag[4];
        #pragma unroll
        for (int nf = 0; nf < 4; ++nf) {
          int row = hl[nf] + r;
          int col = wc[nf] + t;
          bfrag[nf] = *(const v16bf*)(Bhalo + (row * 66 + col) * KC + khalf * 16);
        }
        #pragma unroll
        for (int mf = 0; mf < 4; ++mf) {
          int m = wm * 64 + mf * 16 + l16;
          v16bf afrag = *(const v16bf*)(At + m * KC + khalf * 16);
          #pragma unroll
          for (int nf = 0; nf < 4; ++nf)
            acc[mf][nf] = __builtin_amdgcn_wmma_f32_16x16x32_bf16(
                false, afrag, false, bfrag[nf], (short)0, acc[mf][nf], false, false);
        }
      }
      __syncthreads();                   // reads done before buffer reuse / B restage
    }
  }

  // ---- epilogue: apply scale*demod per cout row, store fp32 ----
  float* ob = out + (size_t)b * OUT_CH * HW;
  const float* osb = oscale + (size_t)b * OUT_CH;
  #pragma unroll
  for (int mf = 0; mf < 4; ++mf) {
    int mbase = m0 + wm * 64 + mf * 16 + khalf * 8;
    #pragma unroll
    for (int nf = 0; nf < 4; ++nf) {
      int np = p0 + wn * 64 + nf * 16 + l16;
      #pragma unroll
      for (int r = 0; r < 8; ++r) {
        int cout = mbase + r;
        ob[(size_t)cout * HW + np] = acc[mf][nf][r] * osb[cout];
      }
    }
  }
}

extern "C" void kernel_launch(void* const* d_in, const int* in_sizes, int n_in,
                              void* d_out, int out_size, void* d_ws, size_t ws_size,
                              hipStream_t stream) {
  const float* x      = (const float*)d_in[0];   // [8,512,64,64]
  const float* style  = (const float*)d_in[1];   // [8,512]
  const float* weight = (const float*)d_in[2];   // [512,512,3,3]
  const float* mod_w  = (const float*)d_in[3];   // [512,512]
  float* out = (float*)d_out;

  char*   ws     = (char*)d_ws;
  float*  s      = (float*)ws;                        // 16 KB
  float*  oscale = (float*)(ws + 16 * 1024);          // 16 KB
  bf16_t* wbf    = (bf16_t*)(ws + 32 * 1024);         // ~4.7 MB

  compute_s_kernel<<<16, 256, 0, stream>>>(style, mod_w, s);
  compute_oscale_kernel<<<16, 256, 0, stream>>>(weight, s, oscale);
  transpose_w_kernel<<<(9 * 512 * 512) / 256, 256, 0, stream>>>(weight, wbf);
  conv_wmma_kernel<<<8 * 2 * 32, 256, 0, stream>>>(x, wbf, s, oscale, out);
}
